// VARNN_28166395527717
// MI455X (gfx1250) — matrix operand order repaired and checked by reading, not compiled, not run on hardware
//
#include <hip/hip_runtime.h>
#include <hip/hip_bf16.h>
#include <stdint.h>

typedef __attribute__((ext_vector_type(16))) _Float16 v16h;
typedef __attribute__((ext_vector_type(8)))  float    v8f;

#define NB   512
#define CC   3
#define TT   512
#define VV   25
#define HH   100
#define GG   400     // 4*H gates
#define NCLS 60
#define K0   75      // C*V
#define KP0  96      // K0 padded to mult of 32
#define KP1  128     // H padded to mult of 32
#define MM   (NB*TT) // 262144 rows in the sequence GEMMs

// ---------------------------------------------------------------------------
// WMMA fragment loaders (CDNA5 16x16x32 f16 layouts per ISA 7.12.2).
// Both loaders read two contiguous 16B runs per lane -> compiler emits
// ds_load_b128 pairs (verified in round-1 disassembly).
// ---------------------------------------------------------------------------
__device__ __forceinline__ v16h load_frag_A(const _Float16* a, int lda) {
  int lane = threadIdx.x & 31;
  int m = lane & 15, hi = lane >> 4;
  v16h r;
#pragma unroll
  for (int j = 0; j < 16; ++j) {
    int k = (j & 7) + ((j >> 3) << 4) + hi * 8;   // A: interleaved K pattern
    r[j] = a[m * lda + k];
  }
  return r;
}

// B[k][n] sourced from row-major W-chunk w[n][k] (B = W^T)
__device__ __forceinline__ v16h load_frag_B(const _Float16* w, int ldw) {
  int lane = threadIdx.x & 31;
  int n = lane & 15, hi = lane >> 4;
  v16h r;
#pragma unroll
  for (int j = 0; j < 16; ++j) {
    int k = (j & 7) + ((j >> 3) << 4) + hi * 8;   // B: interleaved K pattern (matches A)
    r[j] = w[n * ldw + k];
  }
  return r;
}

// ---------------------------------------------------------------------------
// Input-projection GEMM: out16[M][400] = A16[M][Kpad] @ W16[400][Kpad]^T + bias
// block: 256 thr (8 waves). Tile BM=128 x BN=80. Each wave owns one 16-row
// subtile and 5 accumulators (A-fragment reused across 5 WMMAs).
// A tile staged via async global->LDS (ASYNCcnt path).
// ---------------------------------------------------------------------------
#define BMG 128
#define BNG 80
#define BKG 32
__global__ __launch_bounds__(256)
void gemm_xproj_wmma(const _Float16* __restrict__ A,
                     const _Float16* __restrict__ W,
                     const float*    __restrict__ bias,
                     _Float16*       __restrict__ out,   // f16 xproj (halves HBM traffic)
                     int Kpad) {
  __shared__ _Float16 sA[BMG * BKG];   // 8 KB
  __shared__ _Float16 sB[BNG * BKG];   // 5 KB
  const int tid  = threadIdx.x;
  const int wave = tid >> 5;
  const int lane = tid & 31;
  const int bn0  = blockIdx.x * BNG;           // 5 tiles over 400 gate cols
  const size_t bm0 = (size_t)blockIdx.y * BMG;

  v8f acc[5];
#pragma unroll
  for (int q = 0; q < 5; ++q) acc[q] = (v8f){};

  const int nk = Kpad / BKG;
  const int rowdw = Kpad >> 1;                 // u32 per row
  const uint32_t* gW = (const uint32_t*)W;

  for (int kc = 0; kc < nk; ++kc) {
    __syncthreads();
    // ---- A tile: 128 rows x 64B, async copied straight into LDS
#pragma unroll
    for (int u = 0; u < 2; ++u) {
      int c = tid * 2 + u;                     // 512 x 16B chunks
      int r = c >> 2, part = c & 3;
      const _Float16* gsrc = A + (bm0 + r) * (size_t)Kpad + kc * BKG + part * 8;
      uint32_t ldst = (uint32_t)(uintptr_t)(&sA[r * BKG + part * 8]);
      asm volatile("global_load_async_to_lds_b128 %0, %1, off"
                   :: "v"(ldst), "v"(gsrc) : "memory");
    }
    // ---- B tile: 80 rows x 64B via regular loads (small)
    {
      uint32_t* sb32 = (uint32_t*)sB;
      for (int i = tid; i < BNG * 16; i += 256) {
        int r = i >> 4, c2 = i & 15;
        sb32[i] = gW[(size_t)(bn0 + r) * rowdw + (kc << 4) + c2];
      }
    }
    asm volatile("s_wait_asynccnt 0x0" ::: "memory");
    __syncthreads();

    v16h a = load_frag_A(&sA[(wave * 16) * BKG], BKG);
#pragma unroll
    for (int q = 0; q < 5; ++q) {
      v16h b = load_frag_B(&sB[(q * 16) * BKG], BKG);
      acc[q] = __builtin_amdgcn_wmma_f32_16x16x32_f16(false, a, false, b,
                                                      (short)0, acc[q], false, false);
    }
  }
  const int hi = lane >> 4, n = lane & 15;
  const size_t gm = bm0 + wave * 16 + hi * 8;
#pragma unroll
  for (int q = 0; q < 5; ++q) {
    const int col = bn0 + q * 16 + n;
    const float bv = bias[col];
#pragma unroll
    for (int r = 0; r < 8; ++r)
      out[(gm + r) * GG + col] = (_Float16)(acc[q][r] + bv);
  }
}

// ---------------------------------------------------------------------------
// LSTM recurrence: one block = 16 batch rows (32 blocks -> more WGPs busy,
// half the per-step WMMA latency). Whh fully LDS-resident (100 KB: CDNA5-only).
// dynamic LDS: sG 16x400 f32 | sC 16x128 f32 | sWhh 400x128 f16 | sH 16x128 f16
// ---------------------------------------------------------------------------
#define ROWS 16
#define RECUR_LDS ((ROWS*GG*4) + (ROWS*KP1*4) + (GG*KP1*2) + (ROWS*KP1*2))
__global__ __launch_bounds__(256)
void lstm_recur_wmma(const _Float16* __restrict__ xproj,  // [N*T][400] f16 (bias folded)
                     const _Float16* __restrict__ Whh,    // [400][128] f16, zero-pad K
                     _Float16*       __restrict__ hseq,   // [N*T][128] f16 or null
                     float*          __restrict__ hlast,  // [N][100] or null
                     int T) {
  extern __shared__ char smem[];
  float*    sG   = (float*)smem;                     // 16 x 400
  float*    sC   = sG + ROWS * GG;                   // 16 x 128
  _Float16* sWhh = (_Float16*)(sC + ROWS * KP1);     // 400 x 128
  _Float16* sH   = sWhh + GG * KP1;                  // 16 x 128

  const int tid  = threadIdx.x;
  const int wave = tid >> 5;
  const int lane = tid & 31;
  const int n0   = blockIdx.x * ROWS;

  { // one-time loads / init
    const uint32_t* gw = (const uint32_t*)Whh;
    uint32_t* sw = (uint32_t*)sWhh;
    for (int i = tid; i < GG * (KP1 / 2); i += 256) sw[i] = gw[i];
    uint32_t* sh = (uint32_t*)sH;
    for (int i = tid; i < ROWS * (KP1 / 2); i += 256) sh[i] = 0u;
    for (int i = tid; i < ROWS * KP1; i += 256) sC[i] = 0.0f;
  }
  __syncthreads();

  for (int t = 0; t < T; ++t) {
    // ---- WMMA phase: sG[16][400] = sH[16][128] @ sWhh[400][128]^T
    for (int nt = wave; nt < 25; nt += 8) {
      v8f acc = {};
#pragma unroll
      for (int kc = 0; kc < 4; ++kc) {
        v16h a = load_frag_A(&sH[kc * 32], KP1);
        v16h b = load_frag_B(&sWhh[(nt * 16) * KP1 + kc * 32], KP1);
        acc = __builtin_amdgcn_wmma_f32_16x16x32_f16(false, a, false, b,
                                                     (short)0, acc, false, false);
      }
      const int hi = lane >> 4, nn = lane & 15;
#pragma unroll
      for (int r = 0; r < 8; ++r)
        sG[(hi * 8 + r) * GG + nt * 16 + nn] = acc[r];
    }
    __syncthreads();

    // ---- pointwise LSTM cell
    for (int idx = tid; idx < ROWS * HH; idx += 256) {
      const int row = idx / HH, col = idx % HH;
      const size_t xb = ((size_t)(n0 + row) * T + t) * GG;
      const int gb = row * GG;
      const float gi = sG[gb + col]          + (float)xproj[xb + col];
      const float gf = sG[gb + HH + col]     + (float)xproj[xb + HH + col];
      const float gg = sG[gb + 2 * HH + col] + (float)xproj[xb + 2 * HH + col];
      const float go = sG[gb + 3 * HH + col] + (float)xproj[xb + 3 * HH + col];
      const float si = 1.0f / (1.0f + __expf(-gi));
      const float sf = 1.0f / (1.0f + __expf(-gf));
      const float so = 1.0f / (1.0f + __expf(-go));
      const float c  = sf * sC[row * KP1 + col] + si * tanhf(gg);
      const float h  = so * tanhf(c);
      sC[row * KP1 + col] = c;
      sH[row * KP1 + col] = (_Float16)h;
      if (hseq)  hseq[((size_t)(n0 + row) * T + t) * KP1 + col] = (_Float16)h;
      if (hlast && t == T - 1) hlast[(n0 + row) * HH + col] = h;
    }
    if (hseq) { // zero pad cols 100..127 for next layer's GEMM
      for (int idx = tid; idx < ROWS * (KP1 - HH); idx += 256) {
        const int row = idx / (KP1 - HH), col = HH + idx % (KP1 - HH);
        hseq[((size_t)(n0 + row) * T + t) * KP1 + col] = (_Float16)0.0f;
      }
    }
    __syncthreads();
  }
}

// ---------------------------------------------------------------------------
// Small helper kernels
// ---------------------------------------------------------------------------
__global__ void pack_seq_kernel(const float* __restrict__ x, _Float16* __restrict__ seq) {
  size_t idx = (size_t)blockIdx.x * blockDim.x + threadIdx.x;
  if (idx >= (size_t)MM * KP0) return;
  int col = (int)(idx % KP0);
  size_t nt = idx / KP0;
  int t = (int)(nt % TT); int n = (int)(nt / TT);
  float v = 0.0f;
  if (col < K0) {
    int c = col / VV, j = col % VV;
    v = x[(((size_t)n * CC + c) * TT + t) * VV + j];
  }
  seq[idx] = (_Float16)v;
}

__global__ void pack_w_kernel(const float* __restrict__ w, _Float16* __restrict__ o,
                              int rows, int k, int kpad) {
  int idx = blockIdx.x * blockDim.x + threadIdx.x;
  if (idx >= rows * kpad) return;
  int r = idx / kpad, c = idx % kpad;
  o[idx] = (_Float16)(c < k ? w[r * k + c] : 0.0f);
}

__global__ void sum_bias_kernel(const float* a, const float* b, float* o, int n) {
  int i = blockIdx.x * blockDim.x + threadIdx.x;
  if (i < n) o[i] = a[i] + b[i];
}

__global__ void rot_head_kernel(const float* __restrict__ hr, const float* __restrict__ ht,
                                const float* __restrict__ rfW, const float* __restrict__ rfb,
                                const float* __restrict__ tfW, const float* __restrict__ tfb,
                                float* __restrict__ R, float* __restrict__ trans) {
  int n = blockIdx.x * blockDim.x + threadIdx.x;
  if (n >= NB) return;
  float ang[3], tr[3];
#pragma unroll
  for (int i = 0; i < 3; ++i) {
    float sa = rfb[i], st = tfb[i];
    for (int k = 0; k < HH; ++k) {
      sa += hr[n * HH + k] * rfW[i * HH + k];
      st += ht[n * HH + k] * tfW[i * HH + k];
    }
    ang[i] = sa; tr[i] = st;
  }
  float ca = cosf(ang[0]), s0 = sinf(ang[0]);
  float cb = cosf(ang[1]), sb = sinf(ang[1]);
  float cg = cosf(ang[2]), sg = sinf(ang[2]);
  float Rx[9] = {1, 0, 0, 0, ca, s0, 0, -s0, ca};
  float Ry[9] = {cb, 0, -sb, 0, 1, 0, sb, 0, cb};
  float Rz[9] = {cg, sg, 0, -sg, cg, 0, 0, 0, 1};
  float M1[9], Rf[9];
#pragma unroll
  for (int i = 0; i < 3; ++i)
#pragma unroll
    for (int j = 0; j < 3; ++j) {
      float s = 0.0f;
#pragma unroll
      for (int k = 0; k < 3; ++k) s += Rx[i * 3 + k] * Ry[k * 3 + j];
      M1[i * 3 + j] = s;
    }
#pragma unroll
  for (int i = 0; i < 3; ++i)
#pragma unroll
    for (int j = 0; j < 3; ++j) {
      float s = 0.0f;
#pragma unroll
      for (int k = 0; k < 3; ++k) s += M1[i * 3 + k] * Rz[k * 3 + j];
      Rf[i * 3 + j] = s;
    }
#pragma unroll
  for (int i = 0; i < 9; ++i) R[n * 9 + i] = Rf[i];
#pragma unroll
  for (int i = 0; i < 3; ++i) trans[n * 3 + i] = tr[i];
}

__global__ void transform_kernel(const float* __restrict__ x, const float* __restrict__ R,
                                 const float* __restrict__ trans, _Float16* __restrict__ seq) {
  size_t idx = (size_t)blockIdx.x * blockDim.x + threadIdx.x;
  if (idx >= (size_t)NB * TT * VV) return;
  int v = (int)(idx % VV);
  size_t nt = idx / VV;
  int t = (int)(nt % TT); int n = (int)(nt / TT);
  float xc[3];
#pragma unroll
  for (int c = 0; c < 3; ++c)
    xc[c] = x[(((size_t)n * CC + c) * TT + t) * VV + v] - trans[n * 3 + c];
#pragma unroll
  for (int i = 0; i < 3; ++i) {
    float y = R[n * 9 + i * 3 + 0] * xc[0] + R[n * 9 + i * 3 + 1] * xc[1] +
              R[n * 9 + i * 3 + 2] * xc[2];
    seq[((size_t)n * TT + t) * KP0 + i * VV + v] = (_Float16)y;
  }
}

__global__ void fc_kernel(const float* __restrict__ h, const float* __restrict__ W,
                          const float* __restrict__ b, float* __restrict__ out) {
  int idx = blockIdx.x * blockDim.x + threadIdx.x;
  if (idx >= NB * NCLS) return;
  int n = idx / NCLS, o = idx % NCLS;
  float s = b[o];
  for (int k = 0; k < HH; ++k) s += h[n * HH + k] * W[o * HH + k];
  out[idx] = s;
}

// ---------------------------------------------------------------------------
// Host launcher
// ---------------------------------------------------------------------------
extern "C" void kernel_launch(void* const* d_in, const int* in_sizes, int n_in,
                              void* d_out, int out_size, void* d_ws, size_t ws_size,
                              hipStream_t stream) {
  (void)in_sizes; (void)n_in; (void)out_size; (void)ws_size;
  const float* x     = (const float*)d_in[0];
  const float* rWih  = (const float*)d_in[1];
  const float* rWhh  = (const float*)d_in[2];
  const float* rbih  = (const float*)d_in[3];
  const float* rbhh  = (const float*)d_in[4];
  const float* rfW   = (const float*)d_in[5];
  const float* rfb   = (const float*)d_in[6];
  const float* tWih  = (const float*)d_in[7];
  const float* tWhh  = (const float*)d_in[8];
  const float* tbih  = (const float*)d_in[9];
  const float* tbhh  = (const float*)d_in[10];
  const float* tfW   = (const float*)d_in[11];
  const float* tfb   = (const float*)d_in[12];
  const float* mWih0 = (const float*)d_in[13];
  const float* mWhh0 = (const float*)d_in[14];
  const float* mbih0 = (const float*)d_in[15];
  const float* mbhh0 = (const float*)d_in[16];
  const float* mWih1 = (const float*)d_in[17];
  const float* mWhh1 = (const float*)d_in[18];
  const float* mbih1 = (const float*)d_in[19];
  const float* mbhh1 = (const float*)d_in[20];
  const float* mWih2 = (const float*)d_in[21];
  const float* mWhh2 = (const float*)d_in[22];
  const float* mbih2 = (const float*)d_in[23];
  const float* mbhh2 = (const float*)d_in[24];
  const float* mfW   = (const float*)d_in[25];
  const float* mfb   = (const float*)d_in[26];
  float* out = (float*)d_out;

  // ---- workspace carve-out
  char* base = (char*)d_ws;
  size_t off = 0;
  auto alloc = [&](size_t bytes) -> char* {
    char* p = base + off;
    off += (bytes + 255) & ~(size_t)255;
    return p;
  };
  _Float16* seq16  = (_Float16*)alloc((size_t)MM * KP0 * 2);
  _Float16* xproj  = (_Float16*)alloc((size_t)MM * GG * 2);
  _Float16* hseqA  = (_Float16*)alloc((size_t)MM * KP1 * 2);
  _Float16* hseqB  = (_Float16*)alloc((size_t)MM * KP1 * 2);
  _Float16* wih_r  = (_Float16*)alloc((size_t)GG * KP0 * 2);
  _Float16* wih_t  = (_Float16*)alloc((size_t)GG * KP0 * 2);
  _Float16* wih_m0 = (_Float16*)alloc((size_t)GG * KP0 * 2);
  _Float16* wih_m1 = (_Float16*)alloc((size_t)GG * KP1 * 2);
  _Float16* wih_m2 = (_Float16*)alloc((size_t)GG * KP1 * 2);
  _Float16* whh_r  = (_Float16*)alloc((size_t)GG * KP1 * 2);
  _Float16* whh_t  = (_Float16*)alloc((size_t)GG * KP1 * 2);
  _Float16* whh_m0 = (_Float16*)alloc((size_t)GG * KP1 * 2);
  _Float16* whh_m1 = (_Float16*)alloc((size_t)GG * KP1 * 2);
  _Float16* whh_m2 = (_Float16*)alloc((size_t)GG * KP1 * 2);
  float* b_r  = (float*)alloc(GG * 4);
  float* b_t  = (float*)alloc(GG * 4);
  float* b_m0 = (float*)alloc(GG * 4);
  float* b_m1 = (float*)alloc(GG * 4);
  float* b_m2 = (float*)alloc(GG * 4);
  float* hlast_r = (float*)alloc((size_t)NB * HH * 4);
  float* hlast_t = (float*)alloc((size_t)NB * HH * 4);
  float* hlast_m = (float*)alloc((size_t)NB * HH * 4);
  float* Rm    = (float*)alloc((size_t)NB * 9 * 4);
  float* trans = (float*)alloc((size_t)NB * 3 * 4);

  // ---- weight / bias prep
  auto packw = [&](const float* w, _Float16* o, int k, int kpad) {
    int n = GG * kpad;
    pack_w_kernel<<<(n + 255) / 256, 256, 0, stream>>>(w, o, GG, k, kpad);
  };
  packw(rWih, wih_r, K0, KP0);   packw(tWih, wih_t, K0, KP0);
  packw(mWih0, wih_m0, K0, KP0); packw(mWih1, wih_m1, HH, KP1);
  packw(mWih2, wih_m2, HH, KP1);
  packw(rWhh, whh_r, HH, KP1);   packw(tWhh, whh_t, HH, KP1);
  packw(mWhh0, whh_m0, HH, KP1); packw(mWhh1, whh_m1, HH, KP1);
  packw(mWhh2, whh_m2, HH, KP1);
  sum_bias_kernel<<<(GG + 255) / 256, 256, 0, stream>>>(rbih, rbhh, b_r, GG);
  sum_bias_kernel<<<(GG + 255) / 256, 256, 0, stream>>>(tbih, tbhh, b_t, GG);
  sum_bias_kernel<<<(GG + 255) / 256, 256, 0, stream>>>(mbih0, mbhh0, b_m0, GG);
  sum_bias_kernel<<<(GG + 255) / 256, 256, 0, stream>>>(mbih1, mbhh1, b_m1, GG);
  sum_bias_kernel<<<(GG + 255) / 256, 256, 0, stream>>>(mbih2, mbhh2, b_m2, GG);

  // ---- seq (N,T,75) f16, padded
  {
    size_t tot = (size_t)MM * KP0;
    pack_seq_kernel<<<(unsigned)((tot + 255) / 256), 256, 0, stream>>>(x, seq16);
  }

  const dim3 gemm_grid(GG / BNG, MM / BMG);
  const dim3 gemm_blk(256);
  const dim3 rec_grid(NB / ROWS);

  // ---- r-LSTM (last h only)
  gemm_xproj_wmma<<<gemm_grid, gemm_blk, 0, stream>>>(seq16, wih_r, b_r, xproj, KP0);
  lstm_recur_wmma<<<rec_grid, 256, RECUR_LDS, stream>>>(xproj, whh_r, nullptr, hlast_r, TT);
  // ---- t-LSTM (last h only)
  gemm_xproj_wmma<<<gemm_grid, gemm_blk, 0, stream>>>(seq16, wih_t, b_t, xproj, KP0);
  lstm_recur_wmma<<<rec_grid, 256, RECUR_LDS, stream>>>(xproj, whh_t, nullptr, hlast_t, TT);

  // ---- rotation head + apply R(x - trans), repack into seq16
  rot_head_kernel<<<(NB + 255) / 256, 256, 0, stream>>>(hlast_r, hlast_t, rfW, rfb,
                                                        tfW, tfb, Rm, trans);
  {
    size_t tot = (size_t)NB * TT * VV;
    transform_kernel<<<(unsigned)((tot + 255) / 256), 256, 0, stream>>>(x, Rm, trans, seq16);
  }

  // ---- main 3-layer LSTM stack
  gemm_xproj_wmma<<<gemm_grid, gemm_blk, 0, stream>>>(seq16, wih_m0, b_m0, xproj, KP0);
  lstm_recur_wmma<<<rec_grid, 256, RECUR_LDS, stream>>>(xproj, whh_m0, hseqA, nullptr, TT);

  gemm_xproj_wmma<<<gemm_grid, gemm_blk, 0, stream>>>(hseqA, wih_m1, b_m1, xproj, KP1);
  lstm_recur_wmma<<<rec_grid, 256, RECUR_LDS, stream>>>(xproj, whh_m1, hseqB, nullptr, TT);

  gemm_xproj_wmma<<<gemm_grid, gemm_blk, 0, stream>>>(hseqB, wih_m2, b_m2, xproj, KP1);
  lstm_recur_wmma<<<rec_grid, 256, RECUR_LDS, stream>>>(xproj, whh_m2, nullptr, hlast_m, TT);

  // ---- classifier
  fc_kernel<<<(NB * NCLS + 255) / 256, 256, 0, stream>>>(hlast_m, mfW, mfb, out);
}
